// VectorQuantizer_86294482911793
// MI455X (gfx1250) — matrix operand order repaired and checked by reading, not compiled, not run on hardware
//
#include <hip/hip_runtime.h>
#include <hip/hip_bf16.h>
#include <math.h>

// Problem constants (from reference)
#define KC   1024          // codebook size
#define DD   256           // embedding dim
#define BB_  16            // batch
#define TT   2048          // time
#define NTOK (BB_ * TT)    // 32768 flattened tokens
#define OUT_ELEMS (BB_ * DD * TT)  // 8388608

typedef __attribute__((ext_vector_type(16))) __bf16 v16bf;
typedef __attribute__((ext_vector_type(8)))  float  v8f;

union Frag16 { v16bf v; uint4 q[2]; };

typedef __attribute__((address_space(3))) char lds_char;

// ---------------------------------------------------------------------------
// 0) zero scratch accumulators (ws is poisoned by harness; must zero each call)
// ---------------------------------------------------------------------------
__global__ void vq_init(unsigned* __restrict__ counts, float* __restrict__ sse) {
  int i = blockIdx.x * blockDim.x + threadIdx.x;
  if (i < KC) counts[i] = 0u;
  if (i == 0) *sse = 0.0f;
}

// ---------------------------------------------------------------------------
// 1) codebook fp32 -> bf16 (row major, K-contiguous) + half squared norms
// ---------------------------------------------------------------------------
__global__ __launch_bounds__(256)
void vq_prep(const float* __restrict__ cb, __bf16* __restrict__ cbh,
             float* __restrict__ cnorm_half) {
  int k = blockIdx.x;
  int d = threadIdx.x;
  float v = cb[k * DD + d];
  cbh[k * DD + d] = (__bf16)v;
  __shared__ float red[256];
  red[d] = v * v;
  __syncthreads();
  for (int s = 128; s > 0; s >>= 1) {
    if (d < s) red[d] += red[d + s];
    __syncthreads();
  }
  if (d == 0) cnorm_half[k] = 0.5f * red[0];
}

// ---------------------------------------------------------------------------
// 2) distance GEMM via v_wmma_f32_16x16x32_bf16 + fused argmin + histogram.
//    block: 256 threads = 8 waves, 128 tokens; wave handles 16 tokens (M=16)
//    and sweeps all 1024 codes in 16-wide chunks (N=16), K-dim = 256.
//    Codebook chunks are double-buffered into LDS with async global->LDS DMA
//    (ASYNCcnt), shared by all 8 waves -> 8x less L2 traffic than round 1.
// ---------------------------------------------------------------------------
#define SROW  264   // bf16 elems per x-tile LDS row (256 + 8 pad)
#define CBROW 264   // bf16 elems per codebook LDS row (256 + 8 pad)
#define XS_BYTES  (128 * SROW * 2)        // 67584
#define CBUF_BYTES (16 * CBROW * 2)       // 8448 per buffer
#define LDS_BYTES (XS_BYTES + 2 * CBUF_BYTES)

__global__ __launch_bounds__(256)
void vq_argmin(const float* __restrict__ x, const __bf16* __restrict__ cbh,
               const float* __restrict__ cnormh, int* __restrict__ idx,
               unsigned* __restrict__ counts) {
  extern __shared__ __align__(16) char smem_raw[];
  __bf16* xs  = (__bf16*)smem_raw;                       // [128][SROW]
  __bf16* cbs = (__bf16*)(smem_raw + XS_BYTES);          // [2][16][CBROW]
  const unsigned lds_base = (unsigned)(size_t)(lds_char*)smem_raw;
  const unsigned cb_base  = lds_base + XS_BYTES;

  const int n0 = blockIdx.x * 128;     // first flat token of tile
  const int bb = n0 >> 11;             // n0 / TT (tile never crosses batch)
  const int t0 = n0 & (TT - 1);
  const float* xbase = x + (size_t)bb * DD * TT + t0;  // x[bb][d][t0+...]

  // stage x tile: fp32 (B,D,T) -> bf16 LDS [token][d], coalesced over t
  for (int i = threadIdx.x; i < 128 * DD; i += 256) {
    int j = i & 127;       // token within tile
    int d = i >> 7;
    xs[j * SROW + d] = (__bf16)xbase[(size_t)d * TT + j];
  }
  __syncthreads();

  const int lane = threadIdx.x & 31;
  const int wv   = threadIdx.x >> 5;   // wave id: tokens wv*16 .. wv*16+15
  const int lm   = lane & 15;
  const int lh   = lane >> 4;

  // async prefetch of one 16-code chunk (8 KB) into LDS buffer `buf`:
  // 512 x 16B segments, 2 per thread, per-lane LDS dest + global addr
  const int seg0 = threadIdx.x * 2;
  const int prow0 = seg0 >> 5, pcol0 = seg0 & 31;
  const int prow1 = (seg0 + 1) >> 5, pcol1 = (seg0 + 1) & 31;
#define PREFETCH_CB(kc_, buf_)                                                \
  do {                                                                        \
    unsigned l0 = cb_base + (unsigned)(buf_) * CBUF_BYTES +                   \
                  (unsigned)(prow0 * CBROW * 2 + pcol0 * 16);                 \
    unsigned l1 = cb_base + (unsigned)(buf_) * CBUF_BYTES +                   \
                  (unsigned)(prow1 * CBROW * 2 + pcol1 * 16);                 \
    const __bf16* g0 = cbh + ((size_t)((kc_)*16 + prow0) * DD) + pcol0 * 8;   \
    const __bf16* g1 = cbh + ((size_t)((kc_)*16 + prow1) * DD) + pcol1 * 8;   \
    asm volatile("global_load_async_to_lds_b128 %0, %1, off"                  \
                 :: "v"(l0), "v"(g0) : "memory");                             \
    asm volatile("global_load_async_to_lds_b128 %0, %1, off"                  \
                 :: "v"(l1), "v"(g1) : "memory");                             \
  } while (0)

  PREFETCH_CB(0, 0);

  // A fragments: lane lm = token row; lanes 0-15 hold K {0..7,16..23},
  // lanes 16-31 hold K {8..15,24..31} of each 32-wide d-chunk (ISA 7.12.2)
  const __bf16* arow = xs + (wv * 16 + lm) * SROW;
  Frag16 a[8];
#pragma unroll
  for (int c = 0; c < 8; ++c) {
    const __bf16* p = arow + 32 * c + lh * 8;
    a[c].q[0] = *(const uint4*)(p);
    a[c].q[1] = *(const uint4*)(p + 16);
  }

  float bestv[8];
  int   besti[8];
#pragma unroll
  for (int r = 0; r < 8; ++r) { bestv[r] = -3.4e38f; besti[r] = 0; }

  for (int kc = 0; kc < KC / 16; ++kc) {
    const int buf = kc & 1;
    const int code0 = kc * 16;
    // my async loads into buf done -> barrier -> everyone's loads done
    asm volatile("s_wait_asynccnt 0" ::: "memory");
    __syncthreads();
    if (kc + 1 < KC / 16) PREFETCH_CB(kc + 1, buf ^ 1);

    // B fragments from LDS: lane lm = code column; lanes 0-15 K {0..7,16..23}, lanes 16-31 K {8..15,24..31}
    const __bf16* brow = cbs + buf * (16 * CBROW) + lm * CBROW + lh * 8;
    Frag16 b[8];
#pragma unroll
    for (int c = 0; c < 8; ++c) {
      const __bf16* p = brow + 32 * c;
      b[c].q[0] = *(const uint4*)(p);
      b[c].q[1] = *(const uint4*)(p + 16);
    }

    // two accumulator chains (K 0..127 and 128..255) for WMMA ILP
    v8f acc0 = {}, acc1 = {};
#pragma unroll
    for (int c = 0; c < 4; ++c) {
      acc0 = __builtin_amdgcn_wmma_f32_16x16x32_bf16(
          false, a[c].v, false, b[c].v, (short)0, acc0, false, false);
      acc1 = __builtin_amdgcn_wmma_f32_16x16x32_bf16(
          false, a[c + 4].v, false, b[c + 4].v, (short)0, acc1, false, false);
    }

    // score = x.c - 0.5*||c||^2 ; argmin dist == argmax score.
    // C layout: VGPR r, lane -> (M = r + 8*lh, N = lm): one code per lane.
    float cn = cnormh[code0 + lm];
#pragma unroll
    for (int r = 0; r < 8; ++r) {
      float s = acc0[r] + acc1[r] - cn;
      if (s > bestv[r]) { bestv[r] = s; besti[r] = code0 + lm; }
    }
  }

  // reduce over the 16 code lanes of each half-wave (xor bits 0..3)
#pragma unroll
  for (int r = 0; r < 8; ++r) {
#pragma unroll
    for (int off = 1; off < 16; off <<= 1) {
      float ov = __shfl_xor(bestv[r], off, 32);
      int   oi = __shfl_xor(besti[r], off, 32);
      if (ov > bestv[r] || (ov == bestv[r] && oi < besti[r])) {
        bestv[r] = ov; besti[r] = oi;
      }
    }
  }

  if (lm == 0) {  // lanes 0 and 16 hold results for tokens r and r+8
#pragma unroll
    for (int r = 0; r < 8; ++r) {
      int tok = wv * 16 + r + 8 * lh;
      idx[n0 + tok] = besti[r];
      atomicAdd(&counts[besti[r]], 1u);
    }
  }
}

// ---------------------------------------------------------------------------
// 3) gather fp32 codebook rows -> out (B,D,T) + squared-error reduction
//    one block per (b,d) row, coalesced over t
// ---------------------------------------------------------------------------
__global__ __launch_bounds__(256)
void vq_gather(const float* __restrict__ x, const float* __restrict__ cb,
               const int* __restrict__ idx, float* __restrict__ out,
               float* __restrict__ sse) {
  int bd = blockIdx.x;        // 0 .. B*D-1
  int bb = bd >> 8;           // / DD
  int d  = bd & (DD - 1);
  const size_t rowoff = ((size_t)bb * DD + d) * TT;
  float acc = 0.0f;
  for (int t = threadIdx.x; t < TT; t += 256) {
    int   k  = idx[bb * TT + t];
    float q  = cb[k * DD + d];
    float xv = x[rowoff + t];
    out[rowoff + t] = q;      // straight-through value == quantized
    float df = q - xv;
    acc += df * df;
  }
  __shared__ float red[256];
  red[threadIdx.x] = acc;
  __syncthreads();
  for (int s = 128; s > 0; s >>= 1) {
    if (threadIdx.x < s) red[threadIdx.x] += red[threadIdx.x + s];
    __syncthreads();
  }
  if (threadIdx.x == 0) atomicAdd(sse, red[0]);
}

// ---------------------------------------------------------------------------
// 4) finalize: vq_loss = 1.25 * mse (e_loss == q_loss numerically), perplexity
// ---------------------------------------------------------------------------
__global__ void vq_finalize(const unsigned* __restrict__ counts,
                            const float* __restrict__ sse,
                            float* __restrict__ out2) {
  __shared__ float red[256];
  float acc = 0.0f;
  for (int k = threadIdx.x; k < KC; k += 256) {
    float p = (float)counts[k] * (1.0f / (float)NTOK);
    acc += -p * logf(p + 1e-10f);
  }
  red[threadIdx.x] = acc;
  __syncthreads();
  for (int s = 128; s > 0; s >>= 1) {
    if (threadIdx.x < s) red[threadIdx.x] += red[threadIdx.x + s];
    __syncthreads();
  }
  if (threadIdx.x == 0) {
    out2[0] = 1.25f * (*sse) * (1.0f / ((float)NTOK * (float)DD));  // vq_loss
    out2[1] = expf(red[0]);                                          // perplexity
  }
}

// ---------------------------------------------------------------------------
extern "C" void kernel_launch(void* const* d_in, const int* in_sizes, int n_in,
                              void* d_out, int out_size, void* d_ws, size_t ws_size,
                              hipStream_t stream) {
  const float* x  = (const float*)d_in[0];   // (16, 256, 2048) f32
  const float* cb = (const float*)d_in[1];   // (1024, 256) f32
  float* out = (float*)d_out;                // 8388608 + 2 floats

  char* ws = (char*)d_ws;
  __bf16*   cbh    = (__bf16*)(ws);                   // 524288 B
  float*    cnormh = (float*)(ws + 524288);           // 4096 B
  int*      idx    = (int*)(ws + 528384);             // 131072 B
  unsigned* counts = (unsigned*)(ws + 659456);        // 4096 B
  float*    sse    = (float*)(ws + 663552);           // 4 B

  vq_init<<<4, 256, 0, stream>>>(counts, sse);
  vq_prep<<<KC, 256, 0, stream>>>(cb, cbh, cnormh);
  vq_argmin<<<NTOK / 128, 256, LDS_BYTES, stream>>>(x, cbh, cnormh, idx, counts);
  vq_gather<<<BB_ * DD, 256, 0, stream>>>(x, cb, idx, out, sse);
  vq_finalize<<<1, 256, 0, stream>>>(counts, sse, out + OUT_ELEMS);
}
